// LSTM_39419209842762
// MI455X (gfx1250) — matrix operand order repaired and checked by reading, not compiled, not run on hardware
//
#include <hip/hip_runtime.h>

// ---------------------------------------------------------------------------
// 2-layer LSTM for MI455X (gfx1250): bf16 WMMA GEMMs + fused gate math in a
// single persistent kernel with grid-wide syncs between timesteps.
// ---------------------------------------------------------------------------

typedef __bf16 bf16_t;
typedef __attribute__((ext_vector_type(16))) __bf16 v16bf;
typedef __attribute__((ext_vector_type(8)))  __bf16 v8bf;
typedef __attribute__((ext_vector_type(8)))  float  v8f;

constexpr int BB = 64;      // batch
constexpr int TT = 512;     // time steps
constexpr int II = 1024;    // input features (layer 0)
constexpr int HH = 1024;    // hidden
constexpr int BH = BB * HH; // 65536: per-layer state elements
constexpr int NBLK = 32;    // persistent blocks (32 blk x 8 waves = 256 waves)

// ---------------------------------------------------------------------------
// activations
// ---------------------------------------------------------------------------
__device__ __forceinline__ float fsigmoid(float x) {
  return 1.0f / (1.0f + __expf(-x));
}
__device__ __forceinline__ float ftanh_fast(float x) {
  float e = __expf(-2.0f * fabsf(x));       // in (0,1], no overflow
  float r = (1.0f - e) / (1.0f + e);
  return copysignf(r, x);
}

// ---------------------------------------------------------------------------
// A-fragment load (16x32 bf16, M = lane&15):
//   lanes 0-15 hold K = 0..15 ; lanes 16-31 hold K = 16..31
//   => per lane: two contiguous 16B chunks at K offsets kb and kb+8
// ---------------------------------------------------------------------------
__device__ __forceinline__ v16bf load_a_frag(const bf16_t* p) {
  v8bf lo = *(const v8bf*)(p);
  v8bf hi = *(const v8bf*)(p + 8);
  return __builtin_shufflevector(lo, hi, 0, 1, 2, 3, 4, 5, 6, 7,
                                 8, 9, 10, 11, 12, 13, 14, 15);
}

// ---------------------------------------------------------------------------
// Grid-wide sync: monotonic counter, release/acquire fences, s_sleep spin.
// ---------------------------------------------------------------------------
__device__ __forceinline__ void grid_sync(unsigned* cnt, unsigned target) {
  __syncthreads();
  __threadfence();                      // release: my writes visible at device
  if (threadIdx.x == 0) {
    atomicAdd(cnt, 1u);
    while (atomicAdd(cnt, 0u) < target) __builtin_amdgcn_s_sleep(1);
  }
  __syncthreads();
  __threadfence();                      // acquire: drop stale cached lines
}

// ---------------------------------------------------------------------------
// One LSTM layer, one timestep, for this wave's (m0: 16 batch rows,
// u0: 16 hidden units, all 4 gates) tile.
//   z = aIn @ Wi^T + aH @ Wh^T   (biases folded in afterwards)
// Weights are row-major [4096, 1024] bf16 -> K-contiguous B operand.
// ---------------------------------------------------------------------------
template <bool WRITE_OUT>
__device__ __forceinline__ void layer_step(
    const bf16_t* __restrict__ aIn,   // lane-adjusted input-act ptr (k=0)
    const bf16_t* __restrict__ aH,    // lane-adjusted hidden-act ptr (k=0)
    const bf16_t* __restrict__ Wi, const bf16_t* __restrict__ Wh,
    const float* __restrict__ bsl,    // combined bias (bi+bh), 4096 floats
    float* __restrict__ cst_l,        // fp32 cell state [64*1024]
    bf16_t* __restrict__ hW,          // bf16 h write buffer [64*1024]
    float* __restrict__ outp,         // d_out base (layer 1 only)
    float* __restrict__ htp, float* __restrict__ ctp,
    float* __restrict__ gip, float* __restrict__ gfp,
    float* __restrict__ ggp, float* __restrict__ gop,
    int t, int m0, int u0, int lane) {
  const int cl = lane & 15;
  const int hif = lane >> 4;
  const size_t kb16 = (size_t)hif * 16;
  // B-operand row offsets for the 4 gate strips (i, f, g, o)
  const size_t r0 = ((size_t)(u0 + cl) << 10) + kb16;
  const size_t r1 = ((size_t)(u0 + cl + 1024) << 10) + kb16;
  const size_t r2 = ((size_t)(u0 + cl + 2048) << 10) + kb16;
  const size_t r3 = ((size_t)(u0 + cl + 3072) << 10) + kb16;

  v8f acc[4] = {};

  // ---- input-projection GEMM: K over II ----
#pragma unroll 2
  for (int k0 = 0; k0 < II; k0 += 32) {
    v16bf a = load_a_frag(aIn + k0);
    v16bf b0 = *(const v16bf*)(Wi + r0 + k0);
    v16bf b1 = *(const v16bf*)(Wi + r1 + k0);
    v16bf b2 = *(const v16bf*)(Wi + r2 + k0);
    v16bf b3 = *(const v16bf*)(Wi + r3 + k0);
    acc[0] = __builtin_amdgcn_wmma_f32_16x16x32_bf16(false, a, false, b0,
                                                     (short)0, acc[0], false, false);
    acc[1] = __builtin_amdgcn_wmma_f32_16x16x32_bf16(false, a, false, b1,
                                                     (short)0, acc[1], false, false);
    acc[2] = __builtin_amdgcn_wmma_f32_16x16x32_bf16(false, a, false, b2,
                                                     (short)0, acc[2], false, false);
    acc[3] = __builtin_amdgcn_wmma_f32_16x16x32_bf16(false, a, false, b3,
                                                     (short)0, acc[3], false, false);
  }
  // ---- recurrent GEMM: K over HH ----
#pragma unroll 2
  for (int k0 = 0; k0 < HH; k0 += 32) {
    v16bf a = load_a_frag(aH + k0);
    v16bf b0 = *(const v16bf*)(Wh + r0 + k0);
    v16bf b1 = *(const v16bf*)(Wh + r1 + k0);
    v16bf b2 = *(const v16bf*)(Wh + r2 + k0);
    v16bf b3 = *(const v16bf*)(Wh + r3 + k0);
    acc[0] = __builtin_amdgcn_wmma_f32_16x16x32_bf16(false, a, false, b0,
                                                     (short)0, acc[0], false, false);
    acc[1] = __builtin_amdgcn_wmma_f32_16x16x32_bf16(false, a, false, b1,
                                                     (short)0, acc[1], false, false);
    acc[2] = __builtin_amdgcn_wmma_f32_16x16x32_bf16(false, a, false, b2,
                                                     (short)0, acc[2], false, false);
    acc[3] = __builtin_amdgcn_wmma_f32_16x16x32_bf16(false, a, false, b3,
                                                     (short)0, acc[3], false, false);
  }

  // ---- fused gate math; C/D layout: elem r -> (M = r + 8*(lane>>4), N = lane&15)
  const int u = u0 + cl;
  const float b0s = bsl[u];
  const float b1s = bsl[u + 1024];
  const float b2s = bsl[u + 2048];
  const float b3s = bsl[u + 3072];
  const bool last = (t == TT - 1);
#pragma unroll
  for (int r = 0; r < 8; ++r) {
    const int brow = m0 + r + (hif << 3);
    float ig = fsigmoid(acc[0][r] + b0s);
    float fg = fsigmoid(acc[1][r] + b1s);
    float gv = ftanh_fast(acc[2][r] + b2s);
    float og = fsigmoid(acc[3][r] + b3s);
    const size_t ci = (size_t)brow * HH + (size_t)u;
    float cn = fg * cst_l[ci] + ig * gv;
    float hy = og * ftanh_fast(cn);
    cst_l[ci] = cn;
    hW[ci] = (bf16_t)hy;
    if (WRITE_OUT)
      outp[(size_t)brow * ((size_t)TT * HH) + (size_t)t * HH + (size_t)u] = hy;
    if (last) {
      htp[ci] = hy;
      ctp[ci] = cn;
      gip[ci] = ig;
      gfp[ci] = fg;
      ggp[ci] = gv;
      gop[ci] = og;
    }
  }
}

// ---------------------------------------------------------------------------
// Persistent kernel: 32 blocks x 256 threads (8 waves) = 256 waves.
// wave task: u-strip = gw>>2 (64 strips of 16 units), m-quad = gw&3 (16 rows).
// ---------------------------------------------------------------------------
__global__ __launch_bounds__(256) void lstm_persist(
    const bf16_t* __restrict__ xbf,  // [B][T][I] bf16
    const bf16_t* __restrict__ wi0, const bf16_t* __restrict__ wh0,
    const bf16_t* __restrict__ wi1, const bf16_t* __restrict__ wh1,
    const float* __restrict__ bsum,  // [2][4096] combined biases
    bf16_t* __restrict__ hbf,        // [2 layers][2 bufs][B*H] bf16
    float* __restrict__ cst,         // [2 layers][B*H] fp32
    unsigned* __restrict__ syncc, float* __restrict__ dout) {
  const int tid = threadIdx.x;
  const int lane = tid & 31;
  const int gw = blockIdx.x * 8 + (tid >> 5);  // 0..255
  const int u0 = (gw >> 2) << 4;
  const int m0 = (gw & 3) << 4;

  const size_t kb8 = (size_t)(lane >> 4) * 16;
  const size_t xrow = (size_t)(m0 + (lane & 15)) * ((size_t)TT * II) + kb8;
  const size_t hrow = (size_t)(m0 + (lane & 15)) * (size_t)HH + kb8;

  float* htp = dout + (size_t)BB * TT * HH;   // hT  [L,B,H]
  float* ctp = htp + 2 * (size_t)BH;          // cT  [L,B,H]
  float* gip = ctp + 2 * (size_t)BH;          // input_gate
  float* gfp = gip + 2 * (size_t)BH;          // forget_gate
  float* ggp = gfp + 2 * (size_t)BH;          // cell_gate
  float* gop = ggp + 2 * (size_t)BH;          // output_gate

  unsigned target = 0;
  for (int t = 0; t < TT; ++t) {
    const bf16_t* h0r = hbf + (size_t)(t & 1) * BH;
    bf16_t* h0w = hbf + (size_t)((t + 1) & 1) * BH;
    const bf16_t* h1r = hbf + 2 * (size_t)BH + (size_t)(t & 1) * BH;
    bf16_t* h1w = hbf + 2 * (size_t)BH + (size_t)((t + 1) & 1) * BH;

    // layer 0: input = x_t, hidden = h0(t-1)
    layer_step<false>(xbf + xrow + (size_t)t * II, h0r + hrow, wi0, wh0, bsum,
                      cst, h0w, nullptr, htp, ctp, gip, gfp, ggp, gop, t, m0,
                      u0, lane);
    target += NBLK;
    grid_sync(syncc, target);

    // layer 1: input = h0(t), hidden = h1(t-1); writes out[:,t,:]
    layer_step<true>(h0w + hrow, h1r + hrow, wi1, wh1, bsum + 4096, cst + BH,
                     h1w, dout, htp + BH, ctp + BH, gip + BH, gfp + BH,
                     ggp + BH, gop + BH, t, m0, u0, lane);
    target += NBLK;
    grid_sync(syncc, target);
  }
}

// ---------------------------------------------------------------------------
// Preprocessing kernels
// ---------------------------------------------------------------------------
__global__ void f2bf_kernel(const float* __restrict__ s, bf16_t* __restrict__ d,
                            int n) {
  int i = blockIdx.x * blockDim.x + threadIdx.x;
  int st = gridDim.x * blockDim.x;
  for (; i < n; i += st) d[i] = (bf16_t)s[i];
}

__global__ void addv_kernel(const float* __restrict__ a,
                            const float* __restrict__ b, float* __restrict__ o,
                            int n) {
  int i = blockIdx.x * blockDim.x + threadIdx.x;
  if (i < n) o[i] = a[i] + b[i];
}

__global__ void init_state_kernel(float* __restrict__ c,
                                  bf16_t* __restrict__ h,
                                  unsigned* __restrict__ s) {
  int i = blockIdx.x * blockDim.x + threadIdx.x;
  if (i < 2 * BH) c[i] = 0.0f;             // cell state, both layers
  if (i < 4 * BH) h[i] = (bf16_t)0.0f;     // h buffers, 2 layers x 2 bufs
  if (i == 0) *s = 0u;                     // sync counter
}

// ---------------------------------------------------------------------------
// Launcher. Inputs: x, Wi0, bi0, Wh0, bh0, Wi1, bi1, Wh1, bh1 (all fp32).
// Output: out | hT | cT | ig | fg | gg | og (fp32, flat concat).
// ---------------------------------------------------------------------------
extern "C" void kernel_launch(void* const* d_in, const int* in_sizes, int n_in,
                              void* d_out, int out_size, void* d_ws,
                              size_t ws_size, hipStream_t stream) {
  (void)in_sizes; (void)n_in; (void)out_size; (void)ws_size;
  const float* x   = (const float*)d_in[0];
  const float* Wi0 = (const float*)d_in[1];
  const float* bi0 = (const float*)d_in[2];
  const float* Wh0 = (const float*)d_in[3];
  const float* bh0 = (const float*)d_in[4];
  const float* Wi1 = (const float*)d_in[5];
  const float* bi1 = (const float*)d_in[6];
  const float* Wh1 = (const float*)d_in[7];
  const float* bh1 = (const float*)d_in[8];

  constexpr int WSZ = 4096 * 1024;           // weight elements
  constexpr int XSZ = BB * TT * II;          // x elements

  // workspace carve-up (~97.2 MB total, all naturally aligned)
  bf16_t* wi0b = (bf16_t*)d_ws;
  bf16_t* wh0b = wi0b + WSZ;
  bf16_t* wi1b = wh0b + WSZ;
  bf16_t* wh1b = wi1b + WSZ;
  bf16_t* xbf  = wh1b + WSZ;
  bf16_t* hbf  = xbf + XSZ;                  // 4*BH bf16
  float*  bsum = (float*)(hbf + 4 * BH);     // 2*4096 fp32
  float*  cst  = bsum + 2 * 4096;            // 2*BH fp32
  unsigned* sy = (unsigned*)(cst + 2 * BH);

  f2bf_kernel<<<2048, 256, 0, stream>>>(Wi0, wi0b, WSZ);
  f2bf_kernel<<<2048, 256, 0, stream>>>(Wh0, wh0b, WSZ);
  f2bf_kernel<<<2048, 256, 0, stream>>>(Wi1, wi1b, WSZ);
  f2bf_kernel<<<2048, 256, 0, stream>>>(Wh1, wh1b, WSZ);
  f2bf_kernel<<<4096, 256, 0, stream>>>(x, xbf, XSZ);
  addv_kernel<<<16, 256, 0, stream>>>(bi0, bh0, bsum, 4096);
  addv_kernel<<<16, 256, 0, stream>>>(bi1, bh1, bsum + 4096, 4096);
  init_state_kernel<<<(4 * BH + 255) / 256, 256, 0, stream>>>(cst, hbf, sy);

  lstm_persist<<<NBLK, 256, 0, stream>>>(xbf, wi0b, wh0b, wi1b, wh1b, bsum,
                                         hbf, cst, sy, (float*)d_out);
}
